// CRSDBlock_8031588843736
// MI455X (gfx1250) — compile-verified, matrix operand order checked by reading, not run on hardware
//
#include <hip/hip_runtime.h>
#include <hip/hip_bf16.h>
#include <math.h>

// ---------------- problem constants ----------------
#define D_H     1024
#define D_R1    512
#define D_R2    256
#define T_STEPS 8192
#define LAYERS  2
#define ALPHA   0.1f

#define NUM_WG      16
#define ROWS_PER_WG (D_H  / NUM_WG)   // 64 h-rows per workgroup
#define R1_PER_WG   (D_R1 / NUM_WG)   // 32
#define R2_PER_WG   (D_R2 / NUM_WG)   // 16

typedef __bf16 bf16_t;
typedef __bf16        v16bf __attribute__((ext_vector_type(16)));
typedef float         v8f   __attribute__((ext_vector_type(8)));
typedef unsigned int  v8u   __attribute__((ext_vector_type(8)));

// bf16 pair unpack (bf16 stored in low/high 16 bits of a dword)
__device__ __forceinline__ float bflo(unsigned u) {
  union { unsigned x; float f; } c; c.x = u << 16; return c.f;
}
__device__ __forceinline__ float bfhi(unsigned u) {
  union { unsigned x; float f; } c; c.x = u & 0xffff0000u; return c.f;
}

// ---------------------------------------------------------------------------
// f32 -> bf16 conversion (grid-stride)
// ---------------------------------------------------------------------------
__global__ void crsd_cvt_bf16(const float* __restrict__ in,
                              bf16_t* __restrict__ out, int n) {
  int idx = blockIdx.x * blockDim.x + threadIdx.x;
  int stride = gridDim.x * blockDim.x;
  for (int i = idx; i < n; i += stride) out[i] = (bf16_t)in[i];
}

// ---------------------------------------------------------------------------
// zero recurrent state + barrier counter
// ---------------------------------------------------------------------------
__global__ void crsd_init_state(float* hbuf, float* r1, float* r2,
                                unsigned* bar) {
  int tid = threadIdx.x;
  for (int i = tid; i < 2 * D_H; i += 256) hbuf[i] = 0.f;
  for (int i = tid; i < D_R1;    i += 256) r1[i]  = 0.f;
  for (int i = tid; i < D_R2;    i += 256) r2[i]  = 0.f;
  if (tid == 0) *bar = 0u;
}

// ---------------------------------------------------------------------------
// Phase A: Xp[t,i] = bias[i] + sum_j xb[t,j] * wb[i,j]   (bf16 WMMA, f32 acc)
// 2x2 register blocking: one 32x32 output tile per wave (4 WMMAs / k-step on
// 2 A + 2 B fragments -> load:wmma = 1:1).  8 waves per block.
// Fragment layout (16-bit A/B): lane L: idx=L&15, half=L>>4; VGPR v holds the
// bf16 pair at dword kp = 8*(v>>2) + 4*half + (v&3) of the 32-wide k-chunk.
// ---------------------------------------------------------------------------
__global__ void crsd_gemm_wmma(const bf16_t* __restrict__ xb,   // T x D_H
                               const bf16_t* __restrict__ wb,   // D_H x D_H
                               const float*  __restrict__ bias, // D_H
                               float* __restrict__ xp) {        // T x D_H
  const int tid  = threadIdx.x;
  const int wave = tid >> 5;
  const int lane = tid & 31;
  const int tile = blockIdx.x * 8 + wave;  // 32x32 macro-tiles
  const int ti = tile >> 5;                // 1024/32 = 32 i-tiles per row
  const int ii = tile & 31;
  const int t0 = ti << 5;
  const int i0 = ii << 5;

  const int half = lane >> 4;
  const int m    = lane & 15;

  const unsigned* xr0 = (const unsigned*)(xb + (size_t)(t0 + m) * D_H);
  const unsigned* xr1 = (const unsigned*)(xb + (size_t)(t0 + 16 + m) * D_H);
  const unsigned* wr0 = (const unsigned*)(wb + (size_t)(i0 + m) * D_H);
  const unsigned* wr1 = (const unsigned*)(wb + (size_t)(i0 + 16 + m) * D_H);

  v8f c00 = {}, c01 = {}, c10 = {}, c11 = {};
#pragma unroll 2
  for (int j0 = 0; j0 < D_H; j0 += 32) {
    const int base = j0 >> 1;  // dword index into bf16 row
    v8u a0u, a1u, b0u, b1u;
#pragma unroll
    for (int v = 0; v < 8; ++v) {
      const int kp = ((v >> 2) << 3) + (half << 2) + (v & 3);
      a0u[v] = xr0[base + kp];
      a1u[v] = xr1[base + kp];
      b0u[v] = wr0[base + kp];
      b1u[v] = wr1[base + kp];
    }
    v16bf a0 = __builtin_bit_cast(v16bf, a0u);
    v16bf a1 = __builtin_bit_cast(v16bf, a1u);
    v16bf b0 = __builtin_bit_cast(v16bf, b0u);
    v16bf b1 = __builtin_bit_cast(v16bf, b1u);
    c00 = __builtin_amdgcn_wmma_f32_16x16x32_bf16(false, a0, false, b0,
                                                  (short)0, c00, false, false);
    c01 = __builtin_amdgcn_wmma_f32_16x16x32_bf16(false, a0, false, b1,
                                                  (short)0, c01, false, false);
    c10 = __builtin_amdgcn_wmma_f32_16x16x32_bf16(false, a1, false, b0,
                                                  (short)0, c10, false, false);
    c11 = __builtin_amdgcn_wmma_f32_16x16x32_bf16(false, a1, false, b1,
                                                  (short)0, c11, false, false);
  }

  const float bv0 = bias[i0 + m];
  const float bv1 = bias[i0 + 16 + m];
#pragma unroll
  for (int v = 0; v < 8; ++v) {
    const int row0 = t0 + v + (half << 3);   // M = v + 8*half
    const int row1 = row0 + 16;
    xp[(size_t)row0 * D_H + i0 + m]      = c00[v] + bv0;
    xp[(size_t)row0 * D_H + i0 + 16 + m] = c01[v] + bv1;
    xp[(size_t)row1 * D_H + i0 + m]      = c10[v] + bv0;
    xp[(size_t)row1 * D_H + i0 + 16 + m] = c11[v] + bv1;
  }
}

// ---------------------------------------------------------------------------
// Phase B: sequential scan. 16 persistent WGs x 256 threads, global spin
// barrier (agent-scope acquire/release), h double-buffered, state staged in
// LDS.  Recurrent weights stream from L2 in bf16 (5 MB/step, uint4 loads,
// shift/mask unpack) -> half the traffic of f32.
// ---------------------------------------------------------------------------
__global__ void crsd_scan(const float*  __restrict__ xp,   // T x D_H (pre-act)
                          const bf16_t* __restrict__ Whb,  // D_H x D_H
                          const bf16_t* __restrict__ V1b,  // D_H x D_R1
                          const bf16_t* __restrict__ U1b,  // D_R1 x D_H
                          const bf16_t* __restrict__ V2b,  // D_H x D_R2
                          const bf16_t* __restrict__ U2b,  // D_R2 x D_H
                          float* __restrict__ out,         // T x D_H
                          float* hbuf,                     // 2*D_H
                          float* r1buf,                    // D_R1
                          float* r2buf,                    // D_R2
                          unsigned* bar) {
  __shared__ __align__(16) float sh [D_H];
  __shared__ __align__(16) float sr1[D_R1];
  __shared__ __align__(16) float sr2[D_R2];

  const int g   = blockIdx.x;
  const int tid = threadIdx.x;
  unsigned epoch = 0;

  for (int t = 0; t < T_STEPS; ++t) {
    const int par = t & 1;
    const float* hprev = hbuf + ((par ^ 1) * D_H);
    float*       hnew  = hbuf + (par * D_H);

    // stage previous state
    for (int j = tid; j < D_H;  j += 256) sh [j] = hprev[j];
    for (int j = tid; j < D_R1; j += 256) sr1[j] = r1buf[j];
    for (int j = tid; j < D_R2; j += 256) sr2[j] = r2buf[j];
    __syncthreads();

    // ---- part 1: 64 h rows per WG, 4 threads per row ----
    {
      const int row = tid >> 2;
      const int q   = tid & 3;
      const int i   = g * ROWS_PER_WG + row;
      float acc = 0.f;
      {
        const uint4*  w  = (const uint4*)(Whb + (size_t)i * D_H) + q * 32;
        const float4* hv = (const float4*)sh + q * 64;
#pragma unroll 4
        for (int k = 0; k < 32; ++k) {
          uint4 ww = w[k];
          float4 h0 = hv[2 * k], h1 = hv[2 * k + 1];
          acc += bflo(ww.x) * h0.x + bfhi(ww.x) * h0.y +
                 bflo(ww.y) * h0.z + bfhi(ww.y) * h0.w +
                 bflo(ww.z) * h1.x + bfhi(ww.z) * h1.y +
                 bflo(ww.w) * h1.z + bfhi(ww.w) * h1.w;
        }
      }
      {
        const uint4*  w  = (const uint4*)(V1b + (size_t)i * D_R1) + q * 16;
        const float4* rv = (const float4*)sr1 + q * 32;
#pragma unroll 4
        for (int k = 0; k < 16; ++k) {
          uint4 ww = w[k];
          float4 h0 = rv[2 * k], h1 = rv[2 * k + 1];
          acc += bflo(ww.x) * h0.x + bfhi(ww.x) * h0.y +
                 bflo(ww.y) * h0.z + bfhi(ww.y) * h0.w +
                 bflo(ww.z) * h1.x + bfhi(ww.z) * h1.y +
                 bflo(ww.w) * h1.z + bfhi(ww.w) * h1.w;
        }
      }
      {
        const uint4*  w  = (const uint4*)(V2b + (size_t)i * D_R2) + q * 8;
        const float4* rv = (const float4*)sr2 + q * 16;
#pragma unroll 4
        for (int k = 0; k < 8; ++k) {
          uint4 ww = w[k];
          float4 h0 = rv[2 * k], h1 = rv[2 * k + 1];
          acc += bflo(ww.x) * h0.x + bfhi(ww.x) * h0.y +
                 bflo(ww.y) * h0.z + bfhi(ww.y) * h0.w +
                 bflo(ww.z) * h1.x + bfhi(ww.z) * h1.y +
                 bflo(ww.w) * h1.z + bfhi(ww.w) * h1.w;
        }
      }
      acc += __shfl_xor(acc, 1, 4);
      acc += __shfl_xor(acc, 2, 4);
      if (q == 0) {
        float v = tanhf(acc + xp[(size_t)t * D_H + i]);
        hnew[i] = v;
        out[(size_t)t * D_H + i] = v;
      }
    }

    // ---- grid barrier #1 (h' visible everywhere) ----
    __threadfence();
    __syncthreads();
    if (tid == 0) {
      __hip_atomic_fetch_add(bar, 1u, __ATOMIC_ACQ_REL, __HIP_MEMORY_SCOPE_AGENT);
      ++epoch;
      while (__hip_atomic_load(bar, __ATOMIC_ACQUIRE, __HIP_MEMORY_SCOPE_AGENT)
             < epoch * NUM_WG) { __builtin_amdgcn_s_sleep(1); }
    }
    __syncthreads();

    // stage full h' into LDS
    for (int j = tid; j < D_H; j += 256) sh[j] = hnew[j];
    __syncthreads();

    // ---- part 2a: r1 rows (32 per WG, 8 threads per row) ----
    {
      const int row = tid >> 3;
      const int q   = tid & 7;
      const int i   = g * R1_PER_WG + row;
      float acc = 0.f;
      const uint4*  w  = (const uint4*)(U1b + (size_t)i * D_H) + q * 16;
      const float4* hv = (const float4*)sh + q * 32;
#pragma unroll 4
      for (int k = 0; k < 16; ++k) {
        uint4 ww = w[k];
        float4 h0 = hv[2 * k], h1 = hv[2 * k + 1];
        acc += bflo(ww.x) * h0.x + bfhi(ww.x) * h0.y +
               bflo(ww.y) * h0.z + bfhi(ww.y) * h0.w +
               bflo(ww.z) * h1.x + bfhi(ww.z) * h1.y +
               bflo(ww.w) * h1.z + bfhi(ww.w) * h1.w;
      }
      acc += __shfl_xor(acc, 1, 8);
      acc += __shfl_xor(acc, 2, 8);
      acc += __shfl_xor(acc, 4, 8);
      if (q == 0) r1buf[i] = (1.f - ALPHA) * sr1[i] + ALPHA * tanhf(acc);
    }
    // ---- part 2b: r2 rows (16 per WG, 16 threads per row) ----
    {
      const int row = tid >> 4;
      const int q   = tid & 15;
      const int i   = g * R2_PER_WG + row;
      float acc = 0.f;
      const uint4*  w  = (const uint4*)(U2b + (size_t)i * D_H) + q * 8;
      const float4* hv = (const float4*)sh + q * 16;
#pragma unroll 4
      for (int k = 0; k < 8; ++k) {
        uint4 ww = w[k];
        float4 h0 = hv[2 * k], h1 = hv[2 * k + 1];
        acc += bflo(ww.x) * h0.x + bfhi(ww.x) * h0.y +
               bflo(ww.y) * h0.z + bfhi(ww.y) * h0.w +
               bflo(ww.z) * h1.x + bfhi(ww.z) * h1.y +
               bflo(ww.w) * h1.z + bfhi(ww.w) * h1.w;
      }
      acc += __shfl_xor(acc, 1, 16);
      acc += __shfl_xor(acc, 2, 16);
      acc += __shfl_xor(acc, 4, 16);
      acc += __shfl_xor(acc, 8, 16);
      if (q == 0) r2buf[i] = (1.f - ALPHA) * sr2[i] + ALPHA * tanhf(acc);
    }

    // ---- grid barrier #2 (r state visible before next step) ----
    __threadfence();
    __syncthreads();
    if (tid == 0) {
      __hip_atomic_fetch_add(bar, 1u, __ATOMIC_ACQ_REL, __HIP_MEMORY_SCOPE_AGENT);
      ++epoch;
      while (__hip_atomic_load(bar, __ATOMIC_ACQUIRE, __HIP_MEMORY_SCOPE_AGENT)
             < epoch * NUM_WG) { __builtin_amdgcn_s_sleep(1); }
    }
    __syncthreads();
  }
}

// ---------------------------------------------------------------------------
extern "C" void kernel_launch(void* const* d_in, const int* in_sizes, int n_in,
                              void* d_out, int out_size, void* d_ws,
                              size_t ws_size, hipStream_t stream) {
  (void)in_sizes; (void)n_in; (void)out_size; (void)ws_size;

  const float* x_seq = (const float*)d_in[0];
  const float* Wx    = (const float*)d_in[1];
  const float* Wh    = (const float*)d_in[2];
  const float* bias  = (const float*)d_in[3];
  const float* V1    = (const float*)d_in[4];
  const float* U1    = (const float*)d_in[5];
  const float* V2    = (const float*)d_in[6];
  const float* U2    = (const float*)d_in[7];
  float* out = (float*)d_out;

  // workspace layout (bytes)
  char* ws = (char*)d_ws;
  size_t off = 0;
  float*  xp  = (float*)(ws + off);  off += (size_t)T_STEPS * D_H * 4;  // 32MB
  bf16_t* xb  = (bf16_t*)(ws + off); off += (size_t)T_STEPS * D_H * 2;  // 16MB
  bf16_t* wxb = (bf16_t*)(ws + off); off += (size_t)D_H * D_H * 2;      //  2MB
  bf16_t* whb = (bf16_t*)(ws + off); off += (size_t)D_H * D_H * 2;      //  2MB
  bf16_t* v1b = (bf16_t*)(ws + off); off += (size_t)D_H * D_R1 * 2;     //  1MB
  bf16_t* u1b = (bf16_t*)(ws + off); off += (size_t)D_R1 * D_H * 2;     //  1MB
  bf16_t* v2b = (bf16_t*)(ws + off); off += (size_t)D_H * D_R2 * 2;     // .5MB
  bf16_t* u2b = (bf16_t*)(ws + off); off += (size_t)D_R2 * D_H * 2;     // .5MB
  float* hbuf  = (float*)(ws + off); off += (size_t)2 * D_H * 4;
  float* r1buf = (float*)(ws + off); off += (size_t)D_R1 * 4;
  float* r2buf = (float*)(ws + off); off += (size_t)D_R2 * 4;
  unsigned* bar = (unsigned*)(ws + off);

  const int gemm_blocks = (T_STEPS / 32) * (D_H / 32) / 8;  // 1024

  for (int l = 0; l < LAYERS; ++l) {
    const float* in_seq = (l == 0) ? x_seq : out;

    crsd_cvt_bf16<<<512, 256, 0, stream>>>(in_seq, xb, T_STEPS * D_H);
    crsd_cvt_bf16<<<256, 256, 0, stream>>>(Wx + (size_t)l * D_H * D_H, wxb,
                                           D_H * D_H);
    crsd_cvt_bf16<<<256, 256, 0, stream>>>(Wh + (size_t)l * D_H * D_H, whb,
                                           D_H * D_H);
    crsd_cvt_bf16<<<128, 256, 0, stream>>>(V1 + (size_t)l * D_H * D_R1, v1b,
                                           D_H * D_R1);
    crsd_cvt_bf16<<<128, 256, 0, stream>>>(U1 + (size_t)l * D_R1 * D_H, u1b,
                                           D_R1 * D_H);
    crsd_cvt_bf16<<<64, 256, 0, stream>>>(V2 + (size_t)l * D_H * D_R2, v2b,
                                          D_H * D_R2);
    crsd_cvt_bf16<<<64, 256, 0, stream>>>(U2 + (size_t)l * D_R2 * D_H, u2b,
                                          D_R2 * D_H);
    crsd_init_state<<<1, 256, 0, stream>>>(hbuf, r1buf, r2buf, bar);

    crsd_gemm_wmma<<<gemm_blocks, 256, 0, stream>>>(
        xb, wxb, bias + (size_t)l * D_H, xp);

    crsd_scan<<<NUM_WG, 256, 0, stream>>>(
        xp, whb, v1b, u1b, v2b, u2b,
        out, hbuf, r1buf, r2buf, bar);
  }
}
